// QLoRALayer_63350767616598
// MI455X (gfx1250) — matrix operand order repaired and checked by reading, not compiled, not run on hardware
//
#include <hip/hip_runtime.h>

// QLoRA fused kernel for gfx1250 (MI455X): y = x @ dequant4(W)^T + 2*(x@A^T)@B^T
// bf16 WMMA main GEMM + WMMA rank-16 epilogue, async-to-LDS staging of x,
// in-flight int4 dequant of W.

#define IN_F   4096
#define OUT_F  4096
#define RANK   16
#define GSZ    64
#define M_TOT  8192
#define BM     128
#define BN     128
#define BK     32
#define NKT    (IN_F / BK)   // 128 k-steps
#define SCALING 2.0f

typedef __bf16 bf16;
typedef __attribute__((ext_vector_type(16))) __bf16 v16bf;
typedef __attribute__((ext_vector_type(8)))  __bf16 v8bf;
typedef __attribute__((ext_vector_type(8)))  float  v8f;
typedef __attribute__((ext_vector_type(4)))  float  v4f;
typedef __attribute__((ext_vector_type(4)))  int    v4i;

struct SmemMain {
  float x[2][BM * BK];   // 32 KB, f32 x tiles (double buffered, filled by async copy)
  bf16  w[2][BM * BK];   // 16 KB, dequantized bf16 W tiles [n][k]
};
struct SmemEpi {
  bf16 t[BM * BK];       // 8 KB, t = x @ A^T tile, K-padded 16..31 with zeros
  bf16 b[BN * BK];       // 8 KB, SCALING * lora_B tile, K-padded with zeros
};
union Smem { SmemMain m; SmemEpi e; };

__launch_bounds__(256)
__global__ void qlora_wmma_kernel(const float* __restrict__ x,
                                  const int*   __restrict__ qw,
                                  const float* __restrict__ scales,
                                  const float* __restrict__ lora_A,
                                  const float* __restrict__ lora_B,
                                  float* __restrict__ out)
{
  __shared__ Smem smem;
  const int tid  = threadIdx.x;
  const int lane = tid & 31;
  const int wv   = tid >> 5;        // wave 0..7
  const int half = lane >> 4;       // 0: lanes 0-15, 1: lanes 16-31
  const int lr   = lane & 15;
  const int m0   = blockIdx.y * BM;
  const int n0   = blockIdx.x * BN;

  // wave grid: 4 m-rows x 2 n-cols; each wave owns 32(m) x 64(n)
  const int mrow = wv >> 1;         // 0..3
  const int ncol = wv & 1;          // 0..1

  v8f acc[2][4] = {};               // 2 m-subtiles x 4 n-subtiles of 16x16 f32
  v8f tacc = {};                    // this wave's 16x16 tile of t = x @ A^T (subtile wv)

  // W staging registers (16 int4 codes + group scale), filled ahead of compute
  v4i  wq[4];
  float wscale = 0.0f;
  const int wrow = tid >> 1;        // 0..127 : W row within tile
  const int wkh  = (tid & 1) * 16;  // k half within BK

  // ---- stage x tile (f32) into LDS via CDNA5 async copy (ASYNCcnt) ----
  auto stage_x_async = [&](int s, int kt) {
    const int k0 = kt * BK;
    unsigned ldsbase = (unsigned)(uintptr_t)(&smem.m.x[s][0]);
    #pragma unroll
    for (int p = 0; p < 4; ++p) {
      int li   = p * 256 + tid;         // 1024 x 16B chunks = 16 KB tile
      int row  = li >> 3;
      int colb = (li & 7) * 16;
      unsigned lds = ldsbase + (unsigned)(row * (BK * 4) + colb);
      unsigned go  = (unsigned)(((m0 + row) * IN_F + k0) * 4 + colb);
      asm volatile("global_load_async_to_lds_b128 %0, %1, %2"
                   :: "v"(lds), "v"(go), "s"(x) : "memory");
    }
  };

  // ---- issue global loads for the W tile (k-step kt) ----
  auto stage_w_load = [&](int kt) {
    const int k0 = kt * BK;
    const int* qp = qw + (size_t)(n0 + wrow) * IN_F + k0 + wkh;
    #pragma unroll
    for (int p = 0; p < 4; ++p) wq[p] = *(const v4i*)(qp + p * 4);
    wscale = scales[(size_t)(n0 + wrow) * (IN_F / GSZ) + (k0 >> 6)];
  };

  // ---- dequantize and store W tile to LDS as bf16 [n][k] (2x ds_store_b128) ----
  auto stage_w_store = [&](int s) {
    v8bf o0, o1;
    #pragma unroll
    for (int e = 0; e < 4; ++e) {
      o0[e]     = (bf16)((float)wq[0][e] * wscale);
      o0[4 + e] = (bf16)((float)wq[1][e] * wscale);
      o1[e]     = (bf16)((float)wq[2][e] * wscale);
      o1[4 + e] = (bf16)((float)wq[3][e] * wscale);
    }
    v8bf* dst = (v8bf*)&smem.m.w[s][wrow * BK + wkh];
    dst[0] = o0;
    dst[1] = o1;
  };

  // ---- one K-step of WMMA compute ----
  auto compute = [&](int s, int kt) {
    const int k0 = kt * BK;
    v16bf af[2], bfr[4], la;

    // A fragments (16x32 bf16): lane<16 holds K 0-15, lane>=16 K 16-31
    const int klo_a = half ? 16 : 0;
    #pragma unroll
    for (int i = 0; i < 2; ++i) {
      // rotate subtile order so frag 0 is this wave's t-subtile (== wv)
      int msub = mrow * 2 + ((ncol + i) & 1);
      const float* p = &smem.m.x[s][(msub * 16 + lr) * BK];
      v4f c0 = *(const v4f*)(p + klo_a);
      v4f c1 = *(const v4f*)(p + klo_a + 4);
      v4f c2 = *(const v4f*)(p + klo_a + 8);
      v4f c3 = *(const v4f*)(p + klo_a + 12);
      #pragma unroll
      for (int h = 0; h < 4; ++h) {
        af[i][h]      = (bf16)c0[h];
        af[i][4 + h]  = (bf16)c1[h];
        af[i][8 + h]  = (bf16)c2[h];
        af[i][12 + h] = (bf16)c3[h];
      }
    }

    // B fragments (32x16 bf16): lane<16 holds K 0-15, lane>=16 K 16-31
    const int klo_b = half ? 16 : 0;
    #pragma unroll
    for (int j = 0; j < 4; ++j) {
      int nl = ncol * 64 + j * 16 + lr;
      const v8bf* q = (const v8bf*)&smem.m.w[s][nl * BK + klo_b];
      v8bf lo = q[0], hi = q[1];
      bfr[j] = __builtin_shufflevector(lo, hi, 0,1,2,3,4,5,6,7,8,9,10,11,12,13,14,15);
    }

    // lora_A fragment as B-matrix (r = column n), read from global (256 KB, L2-resident)
    {
      const float* p = lora_A + (size_t)lr * IN_F + k0 + klo_b;
      v4f c0 = *(const v4f*)(p);
      v4f c1 = *(const v4f*)(p + 4);
      v4f c2 = *(const v4f*)(p + 8);
      v4f c3 = *(const v4f*)(p + 12);
      #pragma unroll
      for (int h = 0; h < 4; ++h) {
        la[h]      = (bf16)c0[h];
        la[4 + h]  = (bf16)c1[h];
        la[8 + h]  = (bf16)c2[h];
        la[12 + h] = (bf16)c3[h];
      }
    }

    #pragma unroll
    for (int i = 0; i < 2; ++i)
      #pragma unroll
      for (int j = 0; j < 4; ++j)
        acc[i][j] = __builtin_amdgcn_wmma_f32_16x16x32_bf16(
            false, af[i], false, bfr[j], (short)0, acc[i][j], false, false);

    // t accumulation for this wave's own m-subtile (frag 0 by construction)
    tacc = __builtin_amdgcn_wmma_f32_16x16x32_bf16(
        false, af[0], false, la, (short)0, tacc, false, false);
  };

  // ================= main pipeline =================
  stage_x_async(0, 0);
  stage_w_load(0);
  stage_w_store(0);

  #pragma unroll 1
  for (int kt = 0; kt < NKT; ++kt) {
    const int s = kt & 1;
    asm volatile("s_wait_asynccnt 0" ::: "memory");   // x tile landed in LDS
    __syncthreads();                                  // W tile ds_stores visible too
    if (kt + 1 < NKT) {
      stage_x_async(s ^ 1, kt + 1);                   // prefetch next x tile (async)
      stage_w_load(kt + 1);                           // issue next W global loads
    }
    compute(s, kt);
    if (kt + 1 < NKT) stage_w_store(s ^ 1);           // dequant + ds_store next W tile
  }

  // ======== epilogue on the matrix pipe: acc += (SCALING*B) applied to t =========
  __syncthreads();   // done with main-loop LDS; reuse as epilogue buffers

  {
    v8bf zero8 = {};
    // scatter t (f32 D-layout) into LDS as bf16 [m][k], rows of 32 with k 16..31 zero
    #pragma unroll
    for (int v = 0; v < 8; ++v) {
      int ml = wv * 16 + v + 8 * half;
      smem.e.t[ml * BK + lr] = (bf16)tacc[v];
    }
    // zero-pad t k=16..31 (one b128 per thread covers all 128 rows)
    *(v8bf*)&smem.e.t[(tid >> 1) * BK + 16 + (tid & 1) * 8] = zero8;
    // stage SCALING * lora_B as bf16 [n][k] with zero padding
    {
      int n  = tid >> 1;
      int rh = (tid & 1) * 8;
      const float* bp = lora_B + (size_t)(n0 + n) * RANK + rh;
      v8bf bb;
      #pragma unroll
      for (int e = 0; e < 8; ++e) bb[e] = (bf16)(SCALING * bp[e]);
      *(v8bf*)&smem.e.b[n * BK + rh]      = bb;
      *(v8bf*)&smem.e.b[n * BK + 16 + rh] = zero8;
    }
  }
  __syncthreads();

  // rank-16 update via 8 more WMMAs, then store
  {
    const int klo_a = half ? 16 : 0;
    const int klo_b = half ? 16 : 0;
    v16bf at[2], bl[4];
    #pragma unroll
    for (int i = 0; i < 2; ++i) {
      int msub = mrow * 2 + ((ncol + i) & 1);
      const bf16* p = &smem.e.t[(msub * 16 + lr) * BK];
      v8bf lo = *(const v8bf*)(p + klo_a);
      v8bf hi = *(const v8bf*)(p + klo_a + 8);
      at[i] = __builtin_shufflevector(lo, hi, 0,1,2,3,4,5,6,7,8,9,10,11,12,13,14,15);
    }
    #pragma unroll
    for (int j = 0; j < 4; ++j) {
      int nl = ncol * 64 + j * 16 + lr;
      const bf16* p = &smem.e.b[nl * BK + klo_b];
      v8bf lo = *(const v8bf*)(p);
      v8bf hi = *(const v8bf*)(p + 8);
      bl[j] = __builtin_shufflevector(lo, hi, 0,1,2,3,4,5,6,7,8,9,10,11,12,13,14,15);
    }
    #pragma unroll
    for (int i = 0; i < 2; ++i)
      #pragma unroll
      for (int j = 0; j < 4; ++j)
        acc[i][j] = __builtin_amdgcn_wmma_f32_16x16x32_bf16(
            false, at[i], false, bl[j], (short)0, acc[i][j], false, false);
  }

  // ================= store =================
  #pragma unroll
  for (int i = 0; i < 2; ++i) {
    int msub = mrow * 2 + ((ncol + i) & 1);
    #pragma unroll
    for (int j = 0; j < 4; ++j) {
      int nl = ncol * 64 + j * 16 + lr;
      #pragma unroll
      for (int v = 0; v < 8; ++v) {
        int ml = msub * 16 + v + 8 * half;
        out[(size_t)(m0 + ml) * OUT_F + (n0 + nl)] = acc[i][j][v];
      }
    }
  }
}

extern "C" void kernel_launch(void* const* d_in, const int* in_sizes, int n_in,
                              void* d_out, int out_size, void* d_ws, size_t ws_size,
                              hipStream_t stream) {
  const float* x      = (const float*)d_in[0];
  const int*   qw     = (const int*)  d_in[1];
  const float* scales = (const float*)d_in[2];
  const float* lora_A = (const float*)d_in[3];
  const float* lora_B = (const float*)d_in[4];
  float* out = (float*)d_out;

  dim3 grid(OUT_F / BN, M_TOT / BM);   // 32 x 64 = 2048 blocks
  qlora_wmma_kernel<<<grid, 256, 0, stream>>>(x, qw, scales, lora_A, lora_B, out);
}
